// SPAIRGlimpseVAE_15470472200215
// MI455X (gfx1250) — compile-verified, matrix operand order checked by reading, not run on hardware
//
#include <hip/hip_runtime.h>
#include <hip/hip_bf16.h>
#include <math.h>

typedef __attribute__((ext_vector_type(16))) _Float16 v16h;
typedef __attribute__((ext_vector_type(8)))  _Float16 v8h;
typedef __attribute__((ext_vector_type(8)))  float    v8f;

#define G_    2048
#define NPG_  512
#define CPG_  128
#define CPG2_ 32

__device__ __forceinline__ float celu1(float x) {
    return x > 0.f ? x : (__expf(x) - 1.f);
}
__device__ __forceinline__ float softplus1(float x) {
    return x > 20.f ? x : log1pf(__expf(x));
}

// One WMMA 16x32 f16 fragment row: lane owns a fixed matrix row (A: M) or
// column (B: N); its 16 f16 elements are two contiguous 16-byte runs at
// k0+h..+7 and k0+16+h..+7 -> two ds_load_b128.
__device__ __forceinline__ v16h ldsFragRow(const _Float16* rowk0, int h8) {
    v8h lo = *(const v8h*)(rowk0 + h8);
    v8h hi = *(const v8h*)(rowk0 + 16 + h8);
    return __builtin_shufflevector(lo, hi, 0, 1, 2, 3, 4, 5, 6, 7,
                                   8, 9, 10, 11, 12, 13, 14, 15);
}

// Stage W[Kvalid x N] (row-major f32, global) into transposed padded f16 LDS
// WT[n][Kpad]. Clamp-index + select: unconditional load, no EXEC branches.
__device__ __forceinline__ void stageWT(_Float16* WT, const float* __restrict__ W,
                                        int N, int Kvalid, int Kpad, int t) {
    for (int i = t; i < N * Kpad; i += 256) {
        const int n = i / Kpad;
        const int k = i - n * Kpad;
        const int kc = (k < Kvalid) ? k : (Kvalid - 1);
        const float val = W[kc * N + n];
        WT[n * Kpad + k] = (_Float16)((k < Kvalid) ? val : 0.f);
    }
}

__global__ __launch_bounds__(256)
void spair_glimpse_fused(
    const float* __restrict__ rgb,  const float* __restrict__ pos,
    const float* __restrict__ pos1, const float* __restrict__ pos2,
    const int*   __restrict__ idx0, const int*   __restrict__ idx1,
    const int*   __restrict__ idx2, const float* __restrict__ eps,
    const float* __restrict__ W1l,  const float* __restrict__ b1l,
    const float* __restrict__ W1g,  const float* __restrict__ b1g,
    const float* __restrict__ W2l,  const float* __restrict__ b2l,
    const float* __restrict__ W2g,  const float* __restrict__ b2g,
    const float* __restrict__ W3l,  const float* __restrict__ b3l,
    const float* __restrict__ W3g,  const float* __restrict__ b3g,
    const float* __restrict__ Wlin, const float* __restrict__ blin,
    float* __restrict__ out)
{
    (void)idx2; // out_index2[p] == p/32 == blockIdx.x by construction

    // ---- persistent LDS (~20KB) ----
    __shared__ float    sPos1[CPG_ * 3];
    __shared__ float    sPos2[CPG2_ * 3];
    __shared__ int      sIdx1[CPG_];
    __shared__ float    cnt1[CPG_];
    __shared__ float    cnt2[CPG2_];
    __shared__ float    gacc[128];
    __shared__ float    f3s[256];
    __shared__ float    outs[256];
    __shared__ alignas(16) _Float16 A2h[CPG2_ * 64];   // segment means, f16
    __shared__ alignas(16) _Float16 X3h[CPG2_ * 160];  // [f2(128), pos2(3), pad]

    // ---- time-multiplexed scratch (40KB) ----
    __shared__ alignas(16) unsigned char sbuf[40 * 1024];
    float*    acc1 = (float*)sbuf;                       // [128*16]  P1-P2
    _Float16* X2h  = (_Float16*)(sbuf + 8 * 1024);       // [128*64]  P2-P3
    _Float16* W2lT = (_Float16*)(sbuf + 24 * 1024);      // [64*64]   P0-P3
    float*    acc2 = (float*)(sbuf + 32 * 1024);         // [32*64]   P0-P4
    _Float16* W2gT = (_Float16*)sbuf;                    // [128*64]  P4-P5
    _Float16* W3lT = (_Float16*)sbuf;                    // [128*160] P6

    const int g    = blockIdx.x;
    const int t    = threadIdx.x;
    const int lane = t & 31;
    const int wave = t >> 5;
    const int fn   = lane & 15;          // fragment row/col within tile
    const int fh   = (lane >> 4) * 8;    // half-wave K-group offset

    // ---- P0: stage geometry + W2l^T, zero accumulators ----
    if (t < 64) {
        __builtin_prefetch(W3l + t * 256, 0, 0);   // 67KB, L2-resident
        __builtin_prefetch(W2g + t * 128, 0, 0);
        __builtin_prefetch(Wlin + t * 1024, 0, 0);
    }
    for (int i = t; i < CPG_ * 3;  i += 256) sPos1[i] = pos1[g * CPG_ * 3 + i];
    for (int i = t; i < CPG2_ * 3; i += 256) sPos2[i] = pos2[g * CPG2_ * 3 + i];
    for (int i = t; i < CPG_; i += 256) {
        sIdx1[i] = idx1[g * CPG_ + i] - g * CPG2_;
        cnt1[i]  = 0.f;
    }
    for (int i = t; i < CPG_ * 16;  i += 256) acc1[i] = 0.f;
    for (int i = t; i < CPG2_ * 64; i += 256) acc2[i] = 0.f;
    if (t < CPG2_) cnt2[t] = 0.f;
    if (t < 128)   gacc[t] = 0.f;
    stageWT(W2lT, W2l, 64, 35, 64, t);
    __syncthreads();

    // ---- P1: conv1 local: per-point [rgb,rel] @ W1l (4->16), celu, seg-sum ----
    for (int pp = 0; pp < 2; ++pp) {
        const int pl = t + pp * 256;
        const int p  = g * NPG_ + pl;
        const int c  = idx0[p] - g * CPG_;
        const float in0 = rgb[p];
        const float rx = pos[p * 3 + 0] - sPos1[c * 3 + 0];
        const float ry = pos[p * 3 + 1] - sPos1[c * 3 + 1];
        const float rz = pos[p * 3 + 2] - sPos1[c * 3 + 2];
        atomicAdd(&cnt1[c], 1.f);
#pragma unroll
        for (int j = 0; j < 16; ++j) {
            float h = in0 * W1l[j] + rx * W1l[16 + j] + ry * W1l[32 + j]
                    + rz * W1l[48 + j] + b1l[j];
            atomicAdd(&acc1[c * 16 + j], celu1(h));
        }
    }
    __syncthreads();

    // ---- P2: conv1 global: mean @ W1g (16->32), celu -> X2h; rel + pad ----
    {
        const int c  = t >> 1;
        const int jb = (t & 1) * 16;
        const float inv = 1.f / fmaxf(cnt1[c], 1.f);
        float m[16];
#pragma unroll
        for (int i = 0; i < 16; ++i) m[i] = acc1[c * 16 + i] * inv;
#pragma unroll
        for (int jj = 0; jj < 16; ++jj) {
            const int j = jb + jj;
            float v = b1g[j];
#pragma unroll
            for (int i = 0; i < 16; ++i) v += m[i] * W1g[i * 32 + j];
            X2h[c * 64 + j] = (_Float16)celu1(v);
        }
        if ((t & 1) == 0) {
            const int c2 = sIdx1[c];
            atomicAdd(&cnt2[c2], 1.f);
            X2h[c * 64 + 32] = (_Float16)(sPos1[c * 3 + 0] - sPos2[c2 * 3 + 0]);
            X2h[c * 64 + 33] = (_Float16)(sPos1[c * 3 + 1] - sPos2[c2 * 3 + 1]);
            X2h[c * 64 + 34] = (_Float16)(sPos1[c * 3 + 2] - sPos2[c2 * 3 + 2]);
#pragma unroll
            for (int k = 35; k < 64; ++k) X2h[c * 64 + k] = (_Float16)0.f;
        }
    }
    __syncthreads();

    // ---- P3: conv2 local: [128,64p] x [64p,64] WMMA, celu, scatter->acc2 ----
#pragma unroll
    for (int tt = 0; tt < 4; ++tt) {
        const int tile = wave * 4 + tt;   // 0..31
        const int mt = tile >> 2;         // 0..7
        const int nt = tile & 3;          // 0..3
        v8f acc = {};
#pragma unroll
        for (int ks = 0; ks < 2; ++ks) {
            v16h a = ldsFragRow(X2h  + (mt * 16 + fn) * 64 + ks * 32, fh);
            v16h b = ldsFragRow(W2lT + (nt * 16 + fn) * 64 + ks * 32, fh);
            acc = __builtin_amdgcn_wmma_f32_16x16x32_f16(false, a, false, b,
                                                         (short)0, acc, false, false);
        }
        const int col = nt * 16 + fn;
        const float bias = b2l[col];
#pragma unroll
        for (int v = 0; v < 8; ++v) {
            const int m = mt * 16 + fh + v;
            atomicAdd(&acc2[sIdx1[m] * 64 + col], celu1(acc[v] + bias));
        }
    }
    __syncthreads();

    // ---- P4: segment mean -> A2h (f16); X3h rel/pad; stage W2g^T ----
    for (int i = t; i < CPG2_ * 64; i += 256) {
        const int c2 = i >> 6;
        A2h[i] = (_Float16)(acc2[i] / fmaxf(cnt2[c2], 1.f));
    }
    for (int i = t; i < CPG2_ * 32; i += 256) {
        const int m = i >> 5;
        const int k = 128 + (i & 31);
        float v = (k < 131) ? sPos2[m * 3 + (k - 128)] : 0.f; // centers==0
        X3h[m * 160 + k] = (_Float16)v;
    }
    __syncthreads();                 // acc1/X2h dead -> sbuf[0:16K) reusable
    stageWT(W2gT, W2g, 128, 64, 64, t);
    __syncthreads();

    // ---- P5: conv2 global: [32,64] x [64,128] WMMA, celu -> X3h[:,0:128] ----
#pragma unroll
    for (int tt = 0; tt < 2; ++tt) {
        const int tile = wave * 2 + tt;   // 0..15
        const int mt = tile >> 3;         // 0..1
        const int nt = tile & 7;          // 0..7
        v8f acc = {};
#pragma unroll
        for (int ks = 0; ks < 2; ++ks) {
            v16h a = ldsFragRow(A2h  + (mt * 16 + fn) * 64 + ks * 32, fh);
            v16h b = ldsFragRow(W2gT + (nt * 16 + fn) * 64 + ks * 32, fh);
            acc = __builtin_amdgcn_wmma_f32_16x16x32_f16(false, a, false, b,
                                                         (short)0, acc, false, false);
        }
        const int col = nt * 16 + fn;
        const float bias = b2g[col];
#pragma unroll
        for (int v = 0; v < 8; ++v) {
            const int m = mt * 16 + fh + v;
            X3h[m * 160 + col] = (_Float16)celu1(acc[v] + bias);
        }
    }
    __syncthreads();                 // W2gT/acc2 dead -> sbuf[0:40K) for W3l^T
    stageWT(W3lT, W3l, 128, 131, 160, t);
    __syncthreads();

    // ---- P6: conv3 local: [32,160p] x [160p,128] WMMA, celu, row-mean ----
#pragma unroll
    for (int tt = 0; tt < 2; ++tt) {
        const int tile = wave * 2 + tt;
        const int mt = tile >> 3;
        const int nt = tile & 7;
        v8f acc = {};
#pragma unroll
        for (int ks = 0; ks < 5; ++ks) {
            v16h a = ldsFragRow(X3h  + (mt * 16 + fn) * 160 + ks * 32, fh);
            v16h b = ldsFragRow(W3lT + (nt * 16 + fn) * 160 + ks * 32, fh);
            acc = __builtin_amdgcn_wmma_f32_16x16x32_f16(false, a, false, b,
                                                         (short)0, acc, false, false);
        }
        const int col = nt * 16 + fn;
        const float bias = b3l[col];
        float s = 0.f;
#pragma unroll
        for (int v = 0; v < 8; ++v) s += celu1(acc[v] + bias);
        atomicAdd(&gacc[col], s);    // sums all 32 rows across tiles/halves
    }
    __syncthreads();

    if (t < 128) gacc[t] *= (1.f / 32.f);
    __syncthreads();

    // ---- P7: conv3 global GEMV (128->256) + celu ----
    {
        float v = b3g[t];
        for (int i = 0; i < 128; ++i) v += gacc[i] * W3g[i * 256 + t];
        f3s[t] = celu1(v);
    }
    __syncthreads();

    // ---- linear head (256->256) ----
    {
        float v = blin[t];
        for (int i = 0; i < 256; ++i) v += f3s[i] * Wlin[i * 256 + t];
        outs[t] = v;
    }
    __syncthreads();

    // ---- heads: z_what [G,64] | z_mask [G,64] | mu [G,128] | sigma [G,128] | f [G,256]
    out[G_ * 384 + g * 256 + t] = f3s[t];
    if (t < 128) {
        const float mu = outs[t];
        const float sg = outs[128 + t];
        const float sigma = softplus1(sg);
        const float z = mu + sigma * eps[g * 128 + t];
        out[G_ * 128 + g * 128 + t] = mu;
        out[G_ * 256 + g * 128 + t] = sigma;
        if (t < 64) out[g * 64 + t] = z;
        else        out[G_ * 64 + g * 64 + (t - 64)] = z;
    }
}

extern "C" void kernel_launch(void* const* d_in, const int* in_sizes, int n_in,
                              void* d_out, int out_size, void* d_ws, size_t ws_size,
                              hipStream_t stream) {
    (void)in_sizes; (void)n_in; (void)out_size; (void)d_ws; (void)ws_size;
    const float* rgb  = (const float*)d_in[0];
    const float* pos  = (const float*)d_in[1];
    const float* pos1 = (const float*)d_in[2];
    const float* pos2 = (const float*)d_in[3];
    const int*   idx0 = (const int*)d_in[4];
    const int*   idx1 = (const int*)d_in[5];
    const int*   idx2 = (const int*)d_in[6];
    const float* eps  = (const float*)d_in[7];
    const float* W1l  = (const float*)d_in[8];
    const float* b1l  = (const float*)d_in[9];
    const float* W1g  = (const float*)d_in[10];
    const float* b1g  = (const float*)d_in[11];
    const float* W2l  = (const float*)d_in[12];
    const float* b2l  = (const float*)d_in[13];
    const float* W2g  = (const float*)d_in[14];
    const float* b2g  = (const float*)d_in[15];
    const float* W3l  = (const float*)d_in[16];
    const float* b3l  = (const float*)d_in[17];
    const float* W3g  = (const float*)d_in[18];
    const float* b3g  = (const float*)d_in[19];
    const float* Wlin = (const float*)d_in[20];
    const float* blin = (const float*)d_in[21];

    spair_glimpse_fused<<<G_, 256, 0, stream>>>(
        rgb, pos, pos1, pos2, idx0, idx1, idx2, eps,
        W1l, b1l, W1g, b1g, W2l, b2l, W2g, b2g,
        W3l, b3l, W3g, b3g, Wlin, blin, (float*)d_out);
}
